// P_Attention_36696200577513
// MI455X (gfx1250) — matrix operand order repaired and checked by reading, not compiled, not run on hardware
//
#include <hip/hip_runtime.h>

// CDNA5 (gfx1250, wave32) WMMA implementation of Performer-style linear attention.
// D = A x B + C with v_wmma_f32_16x16x32_f16; all GEMM operands staged as f16,
// accumulation in f32. Omega is folded into WQ/WK. ksum via ones-row trick.

typedef __attribute__((ext_vector_type(16))) _Float16 v16h;
typedef __attribute__((ext_vector_type(8)))  _Float16 v8h;
typedef __attribute__((ext_vector_type(8)))  float    v8f;

union V16U { v16h v; v8h h[2]; };

#define DM     1024      // DMODEL
#define DEPTH  1024
#define NHEAD  16
#define NB     4
#define LSEQ   4096
#define CH     16        // K-dim split chunks for kv reduction
#define EPSF   1e-6f

static __device__ __forceinline__ v16h load_frag16(const _Float16* p0, const _Float16* p1) {
  V16U u;
  u.h[0] = *(const v8h*)p0;
  u.h[1] = *(const v8h*)p1;
  return u.v;
}

static __device__ __forceinline__ v8f wmma_f16(v16h a, v16h b, v8f c) {
  return __builtin_amdgcn_wmma_f32_16x16x32_f16(false, a, false, b, (short)0, c, false, false);
}

// ---------------------------------------------------------------------------
// prep: fp32 -> fp16 activation conversion
__global__ void cvt_f16(const float* __restrict__ src, _Float16* __restrict__ dst, int n) {
  int i = blockIdx.x * blockDim.x + threadIdx.x;
  int stride = gridDim.x * blockDim.x;
  for (; i < n; i += stride) dst[i] = (_Float16)src[i];
}

// prep: weight transpose (to [out][in], f16), optionally folding omega per head:
//   WT[h*64+e][i] = sum_d W[i][h*64+d] * omega[d][e]
__global__ void prep_w(const float* __restrict__ W, const float* __restrict__ omega,
                       _Float16* __restrict__ WT, int fold) {
  int idx = blockIdx.x * blockDim.x + threadIdx.x;
  if (idx >= DM * DEPTH) return;
  int c = idx >> 10;      // output (column of W) index
  int i = idx & 1023;     // input (row of W) index
  float v;
  if (fold) {
    int h = c >> 6, e = c & 63;
    float s = 0.f;
    #pragma unroll 8
    for (int d = 0; d < 64; d++) s += W[(long)i * DEPTH + h * 64 + d] * omega[d * 64 + e];
    v = s;
  } else {
    v = W[(long)i * DEPTH + c];
  }
  WT[(long)c * DM + i] = (_Float16)v;
}

// prep: folded bias  b'[h*64+e] = sum_d b[h*64+d] * omega[d][e]
__global__ void prep_b(const float* __restrict__ b, const float* __restrict__ omega,
                       float* __restrict__ bout) {
  int c = blockIdx.x * blockDim.x + threadIdx.x;
  if (c >= DEPTH) return;
  int h = c >> 6, e = c & 63;
  float s = 0.f;
  #pragma unroll 8
  for (int d = 0; d < 64; d++) s += b[h * 64 + d] * omega[d * 64 + e];
  bout[c] = s;
}

// ---------------------------------------------------------------------------
// Generic f16 WMMA GEMM: C(MxN) = A(MxK) * BT(NxK)^T + bias, wave = 64x64 tile.
// mode 0: f32 row-major out (+bias)          -- final projection
// mode 1: relu+eps, f16 row-major out        -- Q_phi
// mode 2: relu+eps, f16 per-head-transposed  -- K_phi^T  (KT[(n,h,d)][l])
// mode 3: bias only, f16 per-head-transposed -- V^T      (VT[(n,h,e)][l])
__global__ __launch_bounds__(256) void gemm_wmma_f16(
    const _Float16* __restrict__ A, const _Float16* __restrict__ BT,
    const float* __restrict__ bias, void* __restrict__ Cout,
    int M, int N, int K, int mode)
{
  const int wave = blockIdx.x * 8 + (threadIdx.x >> 5);
  const int ntile = N >> 6;
  const int wm = wave / ntile;
  const int wn = wave - wm * ntile;
  if (wm * 64 >= M) return;
  const int lane = threadIdx.x & 31;
  const int lh = lane >> 4;      // lane half (0/1)
  const int lm = lane & 15;

  v8f acc[4][4] = {};
  const _Float16* Abase = A  + (long)(wm * 64 + lm) * K;
  const _Float16* Bbase = BT + (long)(wn * 64 + lm) * K;

  for (int k0 = 0; k0 < K; k0 += 32) {
    v16h af[4], bf[4];
    const int ka = k0 + lh * 8;
    #pragma unroll
    for (int mi = 0; mi < 4; mi++) {
      const _Float16* p = Abase + (long)(mi * 16) * K + ka;
      af[mi] = load_frag16(p, p + 16);
    }
    const int kb = k0 + lh * 8;
    #pragma unroll
    for (int ni = 0; ni < 4; ni++) {
      const _Float16* p = Bbase + (long)(ni * 16) * K + kb;
      bf[ni] = load_frag16(p, p + 16);
    }
    #pragma unroll
    for (int mi = 0; mi < 4; mi++)
      #pragma unroll
      for (int ni = 0; ni < 4; ni++)
        acc[mi][ni] = wmma_f16(af[mi], bf[ni], acc[mi][ni]);
  }

  const bool relu = (mode == 1) || (mode == 2);
  #pragma unroll
  for (int ni = 0; ni < 4; ni++) {
    const int col = wn * 64 + ni * 16 + lm;
    const float bv = bias ? bias[col] : 0.0f;
    #pragma unroll
    for (int mi = 0; mi < 4; mi++) {
      #pragma unroll
      for (int i = 0; i < 8; i++) {
        const int row = wm * 64 + mi * 16 + lh * 8 + i;
        float v = acc[mi][ni][i] + bv;
        if (relu) v = fmaxf(v, 0.0f) + EPSF;
        if (mode == 0) {
          ((float*)Cout)[(long)row * N + col] = v;
        } else if (mode == 1) {
          ((_Float16*)Cout)[(long)row * N + col] = (_Float16)v;
        } else {
          const int nb = row >> 12, l = row & (LSEQ - 1);
          const int h = col >> 6,  d = col & 63;
          ((_Float16*)Cout)[(((long)(nb * NHEAD + h) * 64 + d) << 12) + l] = (_Float16)v;
        }
      }
    }
  }
}

// ---------------------------------------------------------------------------
// kv partials per (n,h,chunk): kvT_ext(80x64) = [VT;ones;0](80xLCH) @ KT(64xLCH)^T
// Row 64 of A is all-ones -> row 64 of C is ksum; rows 65..79 zero.
__global__ __launch_bounds__(256) void kv_partial(
    const _Float16* __restrict__ KT, const _Float16* __restrict__ VT,
    float* __restrict__ kvp)
{
  const int wave = blockIdx.x * 8 + (threadIdx.x >> 5);
  const int nh = wave / CH;
  const int ch = wave - nh * CH;
  if (nh >= NB * NHEAD) return;
  const int lane = threadIdx.x & 31;
  const int lh = lane >> 4;
  const int lm = lane & 15;
  const int LCH = LSEQ / CH;       // 256
  const int lbase = ch * LCH;

  const _Float16* vt = VT + ((long)nh * 64 + lm) * LSEQ + lbase;
  const _Float16* kt = KT + ((long)nh * 64 + lm) * LSEQ + lbase;

  // synthetic A fragment for M-tile 4: row e = 64 + lm; e==64 -> ones
  V16U us;
  const _Float16 fv = (lm == 0) ? (_Float16)1.0f : (_Float16)0.0f;
  #pragma unroll
  for (int j = 0; j < 8; j++) { us.h[0][j] = fv; us.h[1][j] = fv; }
  const v16h aones = us.v;

  v8f acc[5][4] = {};
  for (int k0 = 0; k0 < LCH; k0 += 32) {
    v16h af[5], bf[4];
    const int ka = k0 + lh * 8;
    #pragma unroll
    for (int mi = 0; mi < 4; mi++) {
      const _Float16* p = vt + (long)(mi * 16) * LSEQ + ka;
      af[mi] = load_frag16(p, p + 16);
    }
    af[4] = aones;
    const int kb = k0 + lh * 8;
    #pragma unroll
    for (int ni = 0; ni < 4; ni++) {
      const _Float16* p = kt + (long)(ni * 16) * LSEQ + kb;
      bf[ni] = load_frag16(p, p + 16);
    }
    #pragma unroll
    for (int mi = 0; mi < 5; mi++)
      #pragma unroll
      for (int ni = 0; ni < 4; ni++)
        acc[mi][ni] = wmma_f16(af[mi], bf[ni], acc[mi][ni]);
  }

  float* out = kvp + (long)(nh * CH + ch) * 80 * 64;
  #pragma unroll
  for (int mi = 0; mi < 5; mi++)
    #pragma unroll
    for (int ni = 0; ni < 4; ni++)
      #pragma unroll
      for (int i = 0; i < 8; i++) {
        const int e = mi * 16 + lh * 8 + i;
        const int d = ni * 16 + lm;
        out[e * 64 + d] = acc[mi][ni][i];
      }
}

// reduce chunk partials -> kvT_ext f16 (rows 0..63 = kv^T, row 64 = ksum)
__global__ void kv_reduce(const float* __restrict__ kvp, _Float16* __restrict__ kvt) {
  const int TOT = NB * NHEAD * 80 * 64;
  int idx = blockIdx.x * blockDim.x + threadIdx.x;
  if (idx >= TOT) return;
  const int nh = idx / (80 * 64);
  const int r  = idx - nh * 80 * 64;
  float s = 0.f;
  #pragma unroll
  for (int ch = 0; ch < CH; ch++) s += kvp[(long)(nh * CH + ch) * 80 * 64 + r];
  kvt[idx] = (_Float16)s;
}

// ---------------------------------------------------------------------------
// attn = (Qphi @ kv_ext) with column 64 = denominator; divide and store f16.
__global__ __launch_bounds__(256) void attn_numer(
    const _Float16* __restrict__ Q, const _Float16* __restrict__ kvt,
    _Float16* __restrict__ attn)
{
  const int wave = blockIdx.x * 8 + (threadIdx.x >> 5);
  const int nh = wave >> 6;            // 64 M-tiles of 64 rows per (n,h)
  const int mt = wave & 63;
  if (nh >= NB * NHEAD) return;
  const int nb = nh >> 4, h = nh & 15;
  const int lane = threadIdx.x & 31;
  const int lh = lane >> 4;
  const int lm = lane & 15;

  const _Float16* q  = Q   + ((long)(nb * LSEQ + mt * 64 + lm)) * DEPTH + h * 64;
  const _Float16* bt = kvt + (long)nh * 80 * 64 + (long)lm * 64;

  v8f acc[4][5] = {};
  #pragma unroll
  for (int k0 = 0; k0 < 64; k0 += 32) {
    v16h af[4], bf[5];
    const int ka = k0 + lh * 8;
    #pragma unroll
    for (int mi = 0; mi < 4; mi++) {
      const _Float16* p = q + (long)(mi * 16) * DEPTH + ka;
      af[mi] = load_frag16(p, p + 16);
    }
    const int kb = k0 + lh * 8;
    #pragma unroll
    for (int ni = 0; ni < 5; ni++) {
      const _Float16* p = bt + (long)(ni * 16) * 64 + kb;
      bf[ni] = load_frag16(p, p + 16);
    }
    #pragma unroll
    for (int mi = 0; mi < 4; mi++)
      #pragma unroll
      for (int ni = 0; ni < 5; ni++)
        acc[mi][ni] = wmma_f16(af[mi], bf[ni], acc[mi][ni]);
  }

  // denominator lives in N-tile 4, local column 0 => lanes 0 and 16.
  // ds_bpermute broadcasts lane0 -> lanes 0..15, lane16 -> lanes 16..31.
  const int bidx = (lane & 16) << 2;   // byte index 0 or 64
  #pragma unroll
  for (int mi = 0; mi < 4; mi++) {
    float inv[8];
    #pragma unroll
    for (int i = 0; i < 8; i++) {
      int dv = __builtin_amdgcn_ds_bpermute(bidx, __builtin_bit_cast(int, acc[mi][4][i]));
      inv[i] = 1.0f / __builtin_bit_cast(float, dv);
    }
    #pragma unroll
    for (int ni = 0; ni < 4; ni++) {
      const int col = h * 64 + ni * 16 + lm;
      #pragma unroll
      for (int i = 0; i < 8; i++) {
        const int l = mt * 64 + mi * 16 + lh * 8 + i;
        attn[((long)(nb * LSEQ + l)) * DEPTH + col] = (_Float16)(acc[mi][ni][i] * inv[i]);
      }
    }
  }
}

// ---------------------------------------------------------------------------
extern "C" void kernel_launch(void* const* d_in, const int* in_sizes, int n_in,
                              void* d_out, int out_size, void* d_ws, size_t ws_size,
                              hipStream_t stream)
{
  (void)in_sizes; (void)n_in; (void)out_size; (void)ws_size;
  const float* x  = (const float*)d_in[0];
  const float* y  = (const float*)d_in[1];
  const float* WQ = (const float*)d_in[2];
  const float* bQ = (const float*)d_in[3];
  const float* WK = (const float*)d_in[4];
  const float* bK = (const float*)d_in[5];
  const float* WV = (const float*)d_in[6];
  const float* bV = (const float*)d_in[7];
  const float* WO = (const float*)d_in[8];
  const float* bO = (const float*)d_in[9];
  const float* om = (const float*)d_in[10];
  float* out = (float*)d_out;

  char* ws = (char*)d_ws;
  const long SZ_ACT = (long)NB * LSEQ * DEPTH * 2;       // 33.5 MB per f16 activation
  _Float16* xh = (_Float16*)(ws);
  _Float16* yh = (_Float16*)(ws + SZ_ACT);
  _Float16* wq = (_Float16*)(ws + 2 * SZ_ACT);
  _Float16* wk = wq + (long)DM * DEPTH;
  _Float16* wv = wk + (long)DM * DEPTH;
  _Float16* wo = wv + (long)DM * DEPTH;
  float* bq = (float*)(wo + (long)DM * DEPTH);
  float* bk = bq + DEPTH;
  _Float16* qphi = (_Float16*)(bk + DEPTH);
  _Float16* kt   = qphi + (long)NB * LSEQ * DEPTH;
  _Float16* vt   = kt   + (long)NB * LSEQ * DEPTH;
  float*    kvp  = (float*)(vt + (long)NB * LSEQ * DEPTH);
  _Float16* kvt  = (_Float16*)(kvp + (long)NB * NHEAD * CH * 80 * 64);
  _Float16* attn = xh;   // alias: xh is dead after the Q projection

  const int TOTA = NB * LSEQ * DM;      // 16,777,216
  cvt_f16<<<4096, 256, 0, stream>>>(x, xh, TOTA);
  cvt_f16<<<4096, 256, 0, stream>>>(y, yh, TOTA);
  prep_w<<<DM * DEPTH / 256, 256, 0, stream>>>(WQ, om, wq, 1);
  prep_w<<<DM * DEPTH / 256, 256, 0, stream>>>(WK, om, wk, 1);
  prep_w<<<DM * DEPTH / 256, 256, 0, stream>>>(WV, om, wv, 0);
  prep_w<<<DM * DEPTH / 256, 256, 0, stream>>>(WO, om, wo, 0);
  prep_b<<<4, 256, 0, stream>>>(bQ, om, bq);
  prep_b<<<4, 256, 0, stream>>>(bK, om, bk);

  const int M = NB * LSEQ;                               // 16384
  const int gemmBlocks = (M / 64) * (DEPTH / 64) / 8;    // 512
  gemm_wmma_f16<<<gemmBlocks, 256, 0, stream>>>(xh, wq, bq, qphi, M, DEPTH, DM, 1);
  gemm_wmma_f16<<<gemmBlocks, 256, 0, stream>>>(yh, wk, bk, kt,  M, DEPTH, DM, 2);
  gemm_wmma_f16<<<gemmBlocks, 256, 0, stream>>>(yh, wv, bV, vt,  M, DEPTH, DM, 3);

  kv_partial<<<NB * NHEAD * CH / 8, 256, 0, stream>>>(kt, vt, kvp);
  kv_reduce<<<(NB * NHEAD * 80 * 64 + 255) / 256, 256, 0, stream>>>(kvp, kvt);
  attn_numer<<<NB * NHEAD * (LSEQ / 64) / 8, 256, 0, stream>>>(qphi, kvt, attn);

  gemm_wmma_f16<<<gemmBlocks, 256, 0, stream>>>(attn, wo, bO, out, M, DM, DEPTH, 0);
}
